// GCNStack_13606456394315
// MI455X (gfx1250) — matrix operand order repaired and checked by reading, not compiled, not run on hardware
//
#include <hip/hip_runtime.h>
#include <hip/hip_bf16.h>

typedef __attribute__((ext_vector_type(16))) __bf16 v16bf;
typedef __attribute__((ext_vector_type(8)))  float  v8f;

#define N_NODES  100000
#define N_EDGES  1600000
#define HIDDEN   128
#define N_LAYER  3
#define N_GRAPHS 64
#define SCAN_NB  ((N_NODES + 255) / 256)   // 391 blocks in level-1 scan

// ---------------------------------------------------------------- utilities
__global__ void zero_f32(float* __restrict__ p, int n) {
    int t = blockIdx.x * blockDim.x + threadIdx.x;
    if (t < n) p[t] = 0.0f;
}

__global__ void zero_i32(int* __restrict__ p, int n) {
    int t = blockIdx.x * blockDim.x + threadIdx.x;
    if (t < n) p[t] = 0;
}

// in-degree histogram over real edges (self-loop handled analytically)
__global__ void count_kernel(const int* __restrict__ dst, int* __restrict__ degi) {
    int e = blockIdx.x * blockDim.x + threadIdx.x;
    if (e < N_EDGES) atomicAdd(&degi[dst[e]], 1);
}

// dis[i] = rsqrt(deg[i] + 1)  (self loop => deg >= 1, no zero guard needed)
__global__ void dis_kernel(const int* __restrict__ degi, float* __restrict__ dis) {
    int i = blockIdx.x * blockDim.x + threadIdx.x;
    if (i < N_NODES) dis[i] = rsqrtf((float)degi[i] + 1.0f);
}

// ------------------------------------------------------------ 3-level scan
// level 1: per-256 block exclusive scan + block totals
__global__ void scan1(const int* __restrict__ degi, int* __restrict__ part,
                      int* __restrict__ bsum) {
    __shared__ int sm[256];
    int tid = threadIdx.x;
    int idx = blockIdx.x * 256 + tid;
    int v = (idx < N_NODES) ? degi[idx] : 0;
    sm[tid] = v;
    __syncthreads();
    #pragma unroll
    for (int off = 1; off < 256; off <<= 1) {
        int t = (tid >= off) ? sm[tid - off] : 0;
        __syncthreads();
        sm[tid] += t;
        __syncthreads();
    }
    if (idx < N_NODES) part[idx] = sm[tid] - v;   // exclusive
    if (tid == 255) bsum[blockIdx.x] = sm[tid];   // block total
}

// level 2: single block scans the 391 block totals (exclusive)
__global__ void scan2(const int* __restrict__ bsum, int* __restrict__ boff) {
    __shared__ int sm[512];
    int tid = threadIdx.x;
    int v = (tid < SCAN_NB) ? bsum[tid] : 0;
    sm[tid] = v;
    __syncthreads();
    #pragma unroll
    for (int off = 1; off < 512; off <<= 1) {
        int t = (tid >= off) ? sm[tid - off] : 0;
        __syncthreads();
        sm[tid] += t;
        __syncthreads();
    }
    if (tid < SCAN_NB) boff[tid] = sm[tid] - v;   // exclusive
}

// level 3: combine and init fill cursors
__global__ void scan3(const int* __restrict__ part, const int* __restrict__ boff,
                      int* __restrict__ rowstart, int* __restrict__ cursor) {
    int idx = blockIdx.x * blockDim.x + threadIdx.x;
    if (idx < N_NODES) {
        int r = part[idx] + boff[idx >> 8];
        rowstart[idx] = r;
        cursor[idx]   = r;
    }
}

// CSR bucket fill: store (src, norm) per incoming-edge slot
__global__ void fill_kernel(const int* __restrict__ src, const int* __restrict__ dst,
                            const float* __restrict__ dis, int* __restrict__ cursor,
                            int* __restrict__ colsrc, float* __restrict__ colw) {
    int e = blockIdx.x * blockDim.x + threadIdx.x;
    if (e >= N_EDGES) return;
    int s = src[e];
    int d = dst[e];
    int pos = atomicAdd(&cursor[d], 1);
    colsrc[pos] = s;
    colw[pos]   = dis[s] * dis[d];
}

// ------------------------------------------------- W (fp32 row-major KxN) ->
// bf16 packed in the exact 32x16 B-fragment order:
//   idx = ((ntile*4 + ktile)*32 + lane)*16 + j
//   value = W[(ktile*32 + (lane/16)*16 + j)*128 + ntile*16 + (lane%16)]
__global__ void prepack_w(const float* __restrict__ W, __bf16* __restrict__ Wb) {
    int idx = blockIdx.x * blockDim.x + threadIdx.x;   // 16384 threads
    if (idx >= HIDDEN * HIDDEN) return;
    int j     =  idx        & 15;
    int lane  = (idx >> 4)  & 31;
    int kt    = (idx >> 9)  & 3;
    int nt    = (idx >> 11) & 7;
    int col   = lane & 15;
    int khalf = lane >> 4;
    int k     = kt * 32 + khalf * 16 + j;
    int n     = nt * 16 + col;
    Wb[idx] = (__bf16)W[k * HIDDEN + n];
}

// ------------------------------------------------------------- WMMA GEMM
// T[N,128] = H[N,128] @ W[128,128], bf16 in regs, fp32 accumulate.
__global__ __launch_bounds__(256)
void gemm_bf16(const float* __restrict__ H, const __bf16* __restrict__ Wb,
               float* __restrict__ T) {
    const int lane  = threadIdx.x & 31;
    const int wv    = threadIdx.x >> 5;
    const int row0  = blockIdx.x * 128 + wv * 16;
    const int r     = lane & 15;
    const int khalf = lane >> 4;
    const int arow  = row0 + r;

    v8f acc[8] = {};

    #pragma unroll
    for (int kt = 0; kt < 4; ++kt) {
        // A fragment: elem j -> K = kt*32 + khalf*16 + j
        v16bf a;
        if (arow < N_NODES) {
            const float* ap = H + (size_t)arow * HIDDEN + kt * 32 + khalf * 16;
            float4 p0 = *(const float4*)(ap);
            float4 p1 = *(const float4*)(ap + 4);
            float4 p2 = *(const float4*)(ap + 8);
            float4 p3 = *(const float4*)(ap + 12);
            a[0]=(__bf16)p0.x; a[1]=(__bf16)p0.y; a[2]=(__bf16)p0.z; a[3]=(__bf16)p0.w;
            a[4]=(__bf16)p1.x; a[5]=(__bf16)p1.y; a[6]=(__bf16)p1.z; a[7]=(__bf16)p1.w;
            a[8]=(__bf16)p2.x; a[9]=(__bf16)p2.y; a[10]=(__bf16)p2.z; a[11]=(__bf16)p2.w;
            a[12]=(__bf16)p3.x; a[13]=(__bf16)p3.y; a[14]=(__bf16)p3.z; a[15]=(__bf16)p3.w;
        } else {
            #pragma unroll
            for (int j = 0; j < 16; ++j) a[j] = (__bf16)0.0f;
        }
        #pragma unroll
        for (int nt = 0; nt < 8; ++nt) {
            v16bf b = *(const v16bf*)(Wb + (((nt * 4 + kt) * 32) + lane) * 16);
            acc[nt] = __builtin_amdgcn_wmma_f32_16x16x32_bf16(
                false, a, false, b, (short)0, acc[nt], false, false);
        }
    }

    // D layout: vgpr rr, lane -> row = row0 + rr + khalf*8, col = nt*16 + r
    #pragma unroll
    for (int nt = 0; nt < 8; ++nt) {
        #pragma unroll
        for (int rr = 0; rr < 8; ++rr) {
            int orow = row0 + rr + khalf * 8;
            if (orow < N_NODES) T[(size_t)orow * HIDDEN + nt * 16 + r] = acc[nt][rr];
        }
    }
}

// ------------------------------------------ fused pull-aggregation per layer
// One wave per node: acc = relu(bias + T[i]*dis[i]^2 + sum_j T[src_j]*w_j)
__global__ __launch_bounds__(256)
void gather_layer(const float* __restrict__ T, const int* __restrict__ rowstart,
                  const int* __restrict__ degi, const int* __restrict__ colsrc,
                  const float* __restrict__ colw, const float* __restrict__ dis,
                  const float* __restrict__ bias, float* __restrict__ acc) {
    int t    = blockIdx.x * blockDim.x + threadIdx.x;   // N_NODES*32 threads
    int i    = t >> 5;
    int lane = t & 31;
    if (i >= N_NODES) return;
    int f0 = lane * 4;

    float di = dis[i];
    float s2 = di * di;
    float4 a  = *(const float4*)(bias + f0);
    float4 tv = *(const float4*)(T + (size_t)i * HIDDEN + f0);
    a.x += tv.x * s2; a.y += tv.y * s2; a.z += tv.z * s2; a.w += tv.w * s2;

    int start = rowstart[i];
    int cnt   = degi[i];
    for (int j = 0; j < cnt; ++j) {
        int   s = colsrc[start + j];
        float w = colw[start + j];
        const float4 m = *(const float4*)(T + (size_t)s * HIDDEN + f0);
        a.x += m.x * w; a.y += m.y * w; a.z += m.z * w; a.w += m.w * w;
    }

    a.x = fmaxf(a.x, 0.0f); a.y = fmaxf(a.y, 0.0f);
    a.z = fmaxf(a.z, 0.0f); a.w = fmaxf(a.w, 0.0f);
    *(float4*)(acc + (size_t)i * HIDDEN + f0) = a;
}

// ------------------------------------------------------------------- pooling
__global__ void pool_kernel(const float* __restrict__ h, const int* __restrict__ batch,
                            float* __restrict__ gsum, float* __restrict__ gcnt) {
    int t = blockIdx.x * blockDim.x + threadIdx.x;      // N_NODES*32 threads
    if (t >= N_NODES * 32) return;
    int i  = t >> 5;
    int f0 = (t & 31) * 4;
    int g  = batch[i];
    const float4 v = *(const float4*)(h + (size_t)i * HIDDEN + f0);
    float* sp = gsum + (size_t)g * HIDDEN + f0;
    atomicAdd(sp + 0, v.x);
    atomicAdd(sp + 1, v.y);
    atomicAdd(sp + 2, v.z);
    atomicAdd(sp + 3, v.w);
    if (f0 == 0) atomicAdd(&gcnt[g], 1.0f);
}

__global__ void pool_finalize(const float* __restrict__ gsum, const float* __restrict__ gcnt,
                              float* __restrict__ out) {
    int t = blockIdx.x * blockDim.x + threadIdx.x;      // 8192 threads
    if (t >= N_GRAPHS * HIDDEN) return;
    int g = t / HIDDEN;
    out[t] = gsum[t] / fmaxf(gcnt[g], 1.0f);
}

// ---------------------------------------------------------------- launcher
extern "C" void kernel_launch(void* const* d_in, const int* in_sizes, int n_in,
                              void* d_out, int out_size, void* d_ws, size_t ws_size,
                              hipStream_t stream) {
    const float* x     = (const float*)d_in[0];
    const int*   ei    = (const int*)d_in[1];
    const int*   batch = (const int*)d_in[2];
    const float* Ws    = (const float*)d_in[3];
    const float* bs    = (const float*)d_in[4];
    float*       out   = (float*)d_out;

    const int* src = ei;
    const int* dst = ei + N_EDGES;

    // workspace carve (256B aligned)
    char* w = (char*)d_ws;
    auto carve = [&](size_t bytes) -> void* {
        void* p = (void*)w;
        w += (bytes + 255) & ~(size_t)255;
        return p;
    };
    int*    degi     = (int*)   carve((size_t)N_NODES * 4);
    int*    part     = (int*)   carve((size_t)N_NODES * 4);
    int*    rowstart = (int*)   carve((size_t)N_NODES * 4);
    int*    cursor   = (int*)   carve((size_t)N_NODES * 4);
    int*    bsum     = (int*)   carve(512 * 4);
    int*    boff     = (int*)   carve(512 * 4);
    float*  dis      = (float*) carve((size_t)N_NODES * 4);
    int*    colsrc   = (int*)   carve((size_t)N_EDGES * 4);
    float*  colw     = (float*) carve((size_t)N_EDGES * 4);
    float*  bufA     = (float*) carve((size_t)N_NODES * HIDDEN * 4);
    float*  bufB     = (float*) carve((size_t)N_NODES * HIDDEN * 4);
    float*  bufT     = (float*) carve((size_t)N_NODES * HIDDEN * 4);
    __bf16* Wb       = (__bf16*)carve((size_t)HIDDEN * HIDDEN * 2);
    float*  gsum     = (float*) carve((size_t)N_GRAPHS * HIDDEN * 4);
    float*  gcnt     = (float*) carve((size_t)N_GRAPHS * 4);

    const int TB = 256;
    auto blocks = [&](long long n) { return (unsigned)((n + TB - 1) / TB); };

    // CSR build + normalization (recomputed every call; no cross-call state)
    zero_i32<<<blocks(N_NODES), TB, 0, stream>>>(degi, N_NODES);
    count_kernel<<<blocks(N_EDGES), TB, 0, stream>>>(dst, degi);
    dis_kernel<<<blocks(N_NODES), TB, 0, stream>>>(degi, dis);
    scan1<<<SCAN_NB, 256, 0, stream>>>(degi, part, bsum);
    scan2<<<1, 512, 0, stream>>>(bsum, boff);
    scan3<<<blocks(N_NODES), TB, 0, stream>>>(part, boff, rowstart, cursor);
    fill_kernel<<<blocks(N_EDGES), TB, 0, stream>>>(src, dst, dis, cursor, colsrc, colw);

    const float* hin = x;
    float* accbuf[N_LAYER] = { bufA, bufB, bufA };
    for (int l = 0; l < N_LAYER; ++l) {
        float* acc = accbuf[l];
        prepack_w<<<blocks(HIDDEN * HIDDEN), TB, 0, stream>>>(Ws + (size_t)l * HIDDEN * HIDDEN, Wb);
        gemm_bf16<<<(N_NODES + 127) / 128, TB, 0, stream>>>(hin, Wb, bufT);
        gather_layer<<<blocks((long long)N_NODES * 32), TB, 0, stream>>>(
            bufT, rowstart, degi, colsrc, colw, dis, bs + (size_t)l * HIDDEN, acc);
        hin = acc;
    }

    zero_f32<<<blocks(N_GRAPHS * HIDDEN), TB, 0, stream>>>(gsum, N_GRAPHS * HIDDEN);
    zero_f32<<<blocks(N_GRAPHS), TB, 0, stream>>>(gcnt, N_GRAPHS);
    pool_kernel<<<blocks((long long)N_NODES * 32), TB, 0, stream>>>(hin, batch, gsum, gcnt);
    pool_finalize<<<blocks(N_GRAPHS * HIDDEN), TB, 0, stream>>>(gsum, gcnt, out);
}
